// TemporalTransformer_20057497272326
// MI455X (gfx1250) — matrix operand order repaired and checked by reading, not compiled, not run on hardware
//
#include <hip/hip_runtime.h>
#include <hip/hip_bf16.h>
#include <math.h>

// ---------------------------------------------------------------------------
// TemporalTransformer for gfx1250 (MI455X, wave32, WMMA + TDM)
// DIM=512, DEPTH=4, HEADS=8, HD=64, MLP=2048, T=128, WS=32, tokens=102400
// ---------------------------------------------------------------------------

typedef __attribute__((ext_vector_type(16))) _Float16 v16h_t;
typedef __attribute__((ext_vector_type(8)))  float    v8f_t;
typedef __attribute__((ext_vector_type(4)))  unsigned int u32x4_t;
typedef __attribute__((ext_vector_type(8)))  int      i32x8_t;
typedef __attribute__((ext_vector_type(4)))  int      i32x4_t;

#define TOKENS   102400      // 4*200*128
#define CDIM     512
#define NWIN     3200        // TOKENS / 32
#define NELEM_X  52428800ull // 4*512*200*128

#if defined(__HIP_DEVICE_COMPILE__) && __has_builtin(__builtin_amdgcn_tensor_load_to_lds)
#define USE_TDM 1
#else
#define USE_TDM 0
#endif

// ------------------------------ permute in --------------------------------
// h[((bi*200+n)*128+t)*512+c] = x[((bi*512+c)*200+n)*128+t] + pos[t*512+c]
__global__ __launch_bounds__(256) void k_permute_in(
    const float* __restrict__ x, const float* __restrict__ pos,
    float* __restrict__ h) {
  size_t i = (size_t)blockIdx.x * 256 + threadIdx.x;
  if (i >= NELEM_X) return;
  int c = (int)(i & 511);
  size_t tok = i >> 9;
  int t  = (int)(tok & 127);
  size_t bn = tok >> 7;
  int n  = (int)(bn % 200);
  int bi = (int)(bn / 200);
  size_t xi = (((size_t)(bi * 512 + c) * 200 + n) << 7) + t;
  h[i] = x[xi] + pos[t * 512 + c];
}

// ------------------------------ permute out -------------------------------
__global__ __launch_bounds__(256) void k_permute_out(
    const float* __restrict__ h, float* __restrict__ out) {
  size_t i = (size_t)blockIdx.x * 256 + threadIdx.x;
  if (i >= NELEM_X) return;
  int t = (int)(i & 127);
  size_t r = i >> 7;
  int n = (int)(r % 200); r /= 200;
  int c = (int)(r % 512);
  int bi = (int)(r / 512);
  size_t hi = ((((size_t)(bi * 200 + n) << 7) + t) << 9) + c;
  out[i] = h[hi];
}

// ------------------------------ f32 -> f16 (8/thread, b128 I/O) -----------
// all counts used are multiples of 2048, so no ragged edge
__global__ __launch_bounds__(256) void k_cvt_f16(
    const float* __restrict__ src, _Float16* __restrict__ dst, size_t count) {
  size_t i = ((size_t)blockIdx.x * 256 + threadIdx.x) * 8;
  if (i >= count) return;
  float4 a = *(const float4*)(src + i);
  float4 b = *(const float4*)(src + i + 4);
  union { uint4 u; _Float16 h[8]; } t;
  t.h[0] = (_Float16)a.x; t.h[1] = (_Float16)a.y;
  t.h[2] = (_Float16)a.z; t.h[3] = (_Float16)a.w;
  t.h[4] = (_Float16)b.x; t.h[5] = (_Float16)b.y;
  t.h[6] = (_Float16)b.z; t.h[7] = (_Float16)b.w;
  *(uint4*)(dst + i) = t.u;
}

// ------------------------------ layernorm ---------------------------------
// one wave32 per token; 512 channels -> 16 per lane
__global__ __launch_bounds__(256) void k_layernorm(
    const float* __restrict__ h, const float* __restrict__ g,
    const float* __restrict__ b, _Float16* __restrict__ z) {
  int lane = threadIdx.x & 31;
  int wave = threadIdx.x >> 5;
  size_t tok = (size_t)blockIdx.x * 8 + wave;
  const float* p = h + tok * CDIM;
  float xv[16];
  float s = 0.f;
#pragma unroll
  for (int i = 0; i < 16; ++i) { xv[i] = p[lane + i * 32]; s += xv[i]; }
#pragma unroll
  for (int m = 16; m; m >>= 1) s += __shfl_xor(s, m, 32);
  float mu = s * (1.0f / 512.0f);
  float v = 0.f;
#pragma unroll
  for (int i = 0; i < 16; ++i) { float d = xv[i] - mu; v += d * d; }
#pragma unroll
  for (int m = 16; m; m >>= 1) v += __shfl_xor(v, m, 32);
  float inv = rsqrtf(v * (1.0f / 512.0f) + 1e-5f);
  _Float16* zp = z + tok * CDIM;
#pragma unroll
  for (int i = 0; i < 16; ++i) {
    int c = lane + i * 32;
    zp[c] = (_Float16)((xv[i] - mu) * inv * g[c] + b[c]);
  }
}

// ------------------------------ attention ---------------------------------
// one block per (window, head). q,k,v: 32x64 f32 in LDS. causal mask,
// softmax applied TWICE (faithful quirk: 2nd softmax sees zeros at masked
// positions, so the a@v contraction must run over all 32 columns).
__global__ __launch_bounds__(256) void k_attention(
    const _Float16* __restrict__ qkv, _Float16* __restrict__ out) {
  __shared__ float qs[32][68];
  __shared__ float ks[32][68];
  __shared__ float vs[32][68];
  __shared__ float ss[32][36];
  int tid = threadIdx.x;
  int w = blockIdx.x >> 3;
  int head = blockIdx.x & 7;
  size_t tok0 = (size_t)w * 32;
  size_t hoff = (size_t)head * 64;
  for (int idx = tid; idx < 2048; idx += 256) {
    int r = idx >> 6, d = idx & 63;
    size_t base = (tok0 + r) * 1536 + hoff + d;
    qs[r][d] = (float)qkv[base];
    ks[r][d] = (float)qkv[base + 512];
    vs[r][d] = (float)qkv[base + 1024];
  }
  __syncthreads();
  for (int idx = tid; idx < 1024; idx += 256) {
    int r = idx >> 5, c = idx & 31;
    float acc = -INFINITY;
    if (c <= r) {
      acc = 0.f;
#pragma unroll 8
      for (int d = 0; d < 64; ++d) acc += qs[r][d] * ks[c][d];
      acc *= 0.125f;  // HD^-0.5
    }
    ss[r][c] = acc;
  }
  __syncthreads();
  if (tid < 32) {
    int r = tid;
    float tmp[32];
    // softmax #1 (causal; masked -> 0)
    float m = -INFINITY;
#pragma unroll
    for (int c = 0; c < 32; ++c) if (c <= r) m = fmaxf(m, ss[r][c]);
    float Z = 0.f;
#pragma unroll
    for (int c = 0; c < 32; ++c) {
      float e = (c <= r) ? __expf(ss[r][c] - m) : 0.0f;
      tmp[c] = e; Z += e;
    }
    float iZ = 1.0f / Z;
    // softmax #2 over the probabilities (includes the zeros)
    float m2 = -INFINITY;
#pragma unroll
    for (int c = 0; c < 32; ++c) m2 = fmaxf(m2, tmp[c] * iZ);
    float Z2 = 0.f;
#pragma unroll
    for (int c = 0; c < 32; ++c) {
      float e = __expf(tmp[c] * iZ - m2);
      tmp[c] = e; Z2 += e;
    }
    float iZ2 = 1.0f / Z2;
#pragma unroll
    for (int c = 0; c < 32; ++c) ss[r][c] = tmp[c] * iZ2;
  }
  __syncthreads();
  for (int idx = tid; idx < 2048; idx += 256) {
    int r = idx >> 6, d = idx & 63;
    float acc = 0.f;
#pragma unroll
    for (int c = 0; c < 32; ++c) acc += ss[r][c] * vs[c][d];
    out[(tok0 + r) * CDIM + hoff + d] = (_Float16)acc;
  }
}

// ------------------------------ WMMA GEMM ---------------------------------
// C[M,N] = A16[M,K] @ W16[K,N] (+bias)(gelu)(+=residual f32 | f16 store)
// Block: 256 thr = 8 waves, tile 128(M) x 64(N), BK = 32 (one WMMA K step).
// Wave grid 4x2; each wave owns a 32x32 tile = 2x2 v_wmma_f32_16x16x32_f16.
// A tile staged by the Tensor Data Mover (one DMA per K-step, wave 0 only,
// D# pad feature reproduces the [128][40]-half padded LDS layout), B tile
// staged manually with an LDS transpose. Double-buffered: 1 barrier/step.
template <bool BIAS, bool GELU, bool RESID, bool OUT16>
__global__ __launch_bounds__(256) void k_gemm_wmma(
    const _Float16* __restrict__ A, const _Float16* __restrict__ W,
    const float* __restrict__ bias, float* __restrict__ resid_out,
    _Float16* __restrict__ out16, int K, int N) {
  __shared__ __align__(16) _Float16 Als[2][128][40];  // [buf][m][k] (+pad 8)
  __shared__ __align__(16) _Float16 Bls[2][64][40];   // [buf][n][k] transposed
  const int tid = threadIdx.x;
  const int lane = tid & 31;
  const int wave = tid >> 5;
  const int wm = wave & 3;        // 0..3 -> M offset wm*32
  const int wn = wave >> 2;       // 0..1 -> N offset wn*32
  const size_t bm0 = (size_t)blockIdx.x * 128;
  const int bn0 = blockIdx.y * 64;
  const int nsteps = K >> 5;

  v8f_t c00 = {}, c01 = {}, c10 = {}, c11 = {};

#define STAGE_B(kk, bf)                                                      \
  {                                                                          \
    int row = tid >> 3;                                                      \
    int ch = (tid & 7) << 3;                                                 \
    union { uint4 u; _Float16 hh[8]; } t;                                    \
    t.u = *(const uint4*)(W + (size_t)((kk) + row) * N + bn0 + ch);          \
    _Pragma("unroll")                                                        \
    for (int j = 0; j < 8; ++j) Bls[bf][ch + j][row] = t.hh[j];              \
  }

#if USE_TDM
  // Tensor DMA: load 128x32-half tile (rows stride K) into padded LDS.
  // 6-arg builtin form: (g0 u32x4, g1 i32x8, g2 i32x4, g3 i32x4, i32x8, cpol)
#define STAGE_A(kk, bf)                                                      \
  if (wave == 0) {                                                           \
    uint64_t ga = (uint64_t)(uintptr_t)(A + bm0 * (size_t)K + (kk));         \
    uint32_t la = (uint32_t)(uintptr_t)(&Als[bf][0][0]);                     \
    u32x4_t g0;                                                              \
    g0[0] = 1u;                          /* count=1, load, user mode */      \
    g0[1] = la;                          /* lds_addr (bytes) */              \
    g0[2] = (uint32_t)ga;                /* global_addr[31:0] */             \
    g0[3] = (uint32_t)((ga >> 32) & 0x1FFFFFFu) | (2u << 30); /* +type=2 */  \
    i32x8_t g1;                                                              \
    /* data_size=2B(1<<16) pad_en(1<<20) pad_int=16DW(3<<22) pad=4DW(3<<25)*/\
    g1[0] = (int)0x06D10000u;                                                \
    g1[1] = (int)(((unsigned)K & 0xFFFFu) << 16);  /* tensor_dim0 lo16 */    \
    g1[2] = (int)((((unsigned)K >> 16) & 0xFFFFu) |                          \
                  (((unsigned)TOKENS & 0xFFFFu) << 16)); /* d0 hi, d1 lo */  \
    g1[3] = (int)(((unsigned)TOKENS >> 16) | (32u << 16)); /* d1 hi, tile0 */\
    g1[4] = 128;                         /* tile_dim1=128, tile_dim2=0 */    \
    g1[5] = K;                           /* tensor_dim0_stride lo32 */       \
    g1[6] = 0;                           /* stride hi, dim1_stride lo */     \
    g1[7] = 0;                                                               \
    i32x4_t z4 = {0, 0, 0, 0};                                               \
    i32x8_t z8 = {0, 0, 0, 0, 0, 0, 0, 0};                                   \
    __builtin_amdgcn_tensor_load_to_lds(g0, g1, z4, z4, z8, 0);              \
  }
#define WAIT_A() __builtin_amdgcn_s_wait_tensorcnt(0)
#else
#define STAGE_A(kk, bf)                                                      \
  {                                                                          \
    _Pragma("unroll")                                                        \
    for (int i = 0; i < 2; ++i) {                                            \
      int idx = i * 256 + tid;                                               \
      int row = idx >> 2;                                                    \
      int ch = (idx & 3) << 3;                                               \
      uint4 t = *(const uint4*)(A + (bm0 + row) * (size_t)K + (kk) + ch);    \
      *(uint4*)(&Als[bf][row][ch]) = t;                                      \
    }                                                                        \
  }
#define WAIT_A()
#endif

  // prologue: stage step 0 into buffer 0
  STAGE_A(0, 0);
  STAGE_B(0, 0);

  for (int s = 0; s < nsteps; ++s) {
    const int b = s & 1;
    WAIT_A();            // TDM write of buffer b complete (wave 0)
    __syncthreads();     // publish buffer b; buffer b^1 fully drained
    if (s + 1 < nsteps) {
      STAGE_A((s + 1) << 5, b ^ 1);
      STAGE_B((s + 1) << 5, b ^ 1);
    }
    // ---- build fragments per ISA 16-bit layouts ----
    v16h_t a0, a1, b0, b1;
    {
      const int mrow = lane & 15;
      const int kh = (lane >> 4) << 3;   // K octet selector for A
      const int kq = (lane >> 4) << 3;   // K group selector for B
      const _Float16* p0 = &Als[b][wm * 32 + mrow][0];
      const _Float16* p1 = &Als[b][wm * 32 + 16 + mrow][0];
#pragma unroll
      for (int e = 0; e < 8; ++e) {
        a0[e] = p0[kh + e];  a0[e + 8] = p0[kh + 16 + e];
        a1[e] = p1[kh + e];  a1[e + 8] = p1[kh + 16 + e];
      }
      const _Float16* q0 = &Bls[b][wn * 32 + mrow][kq];
      const _Float16* q1 = &Bls[b][wn * 32 + 16 + mrow][kq];
#pragma unroll
      for (int e = 0; e < 8; ++e) { b0[e] = q0[e]; b0[e + 8] = q0[16 + e]; b1[e] = q1[e]; b1[e + 8] = q1[16 + e]; }
    }
    c00 = __builtin_amdgcn_wmma_f32_16x16x32_f16(false, a0, false, b0, (short)0, c00, false, false);
    c01 = __builtin_amdgcn_wmma_f32_16x16x32_f16(false, a0, false, b1, (short)0, c01, false, false);
    c10 = __builtin_amdgcn_wmma_f32_16x16x32_f16(false, a1, false, b0, (short)0, c10, false, false);
    c11 = __builtin_amdgcn_wmma_f32_16x16x32_f16(false, a1, false, b1, (short)0, c11, false, false);
  }
#undef STAGE_A
#undef STAGE_B
#undef WAIT_A

  // ---- epilogue: C/D layout = lane%16 -> N, vgpr r + 8*(lane/16) -> M ----
  const int lm = (lane >> 4) << 3;
  const int lnn = lane & 15;
#pragma unroll
  for (int tm = 0; tm < 2; ++tm) {
#pragma unroll
    for (int tn = 0; tn < 2; ++tn) {
      v8f_t acc = (tm == 0) ? ((tn == 0) ? c00 : c01)
                            : ((tn == 0) ? c10 : c11);
#pragma unroll
      for (int r = 0; r < 8; ++r) {
        size_t M = bm0 + wm * 32 + tm * 16 + lm + r;
        int Nc = bn0 + wn * 32 + tn * 16 + lnn;
        float v = acc[r];
        if (BIAS) v += bias[Nc];
        if (GELU) v = 0.5f * v * (1.0f + erff(v * 0.70710678118654752f));
        size_t o = M * (size_t)N + Nc;
        if (RESID)      resid_out[o] = resid_out[o] + v;
        else if (OUT16) out16[o] = (_Float16)v;
        else            resid_out[o] = v;
      }
    }
  }
}

// ---------------------------------------------------------------------------
extern "C" void kernel_launch(void* const* d_in, const int* in_sizes, int n_in,
                              void* d_out, int out_size, void* d_ws, size_t ws_size,
                              hipStream_t stream) {
  (void)in_sizes; (void)n_in; (void)out_size; (void)ws_size;
  const float* x      = (const float*)d_in[0];
  const float* pos    = (const float*)d_in[1];
  const float* qkv_w  = (const float*)d_in[2];
  const float* proj_w = (const float*)d_in[3];
  const float* proj_b = (const float*)d_in[4];
  const float* ln_g   = (const float*)d_in[5];
  const float* ln_b   = (const float*)d_in[6];
  const float* ff_w1  = (const float*)d_in[7];
  const float* ff_b1  = (const float*)d_in[8];
  const float* ff_w2  = (const float*)d_in[9];
  const float* ff_b2  = (const float*)d_in[10];
  float* out = (float*)d_out;

  // workspace layout (bytes):
  //   [0, 209715200)            h      : residual stream f32 (102400x512)
  //   [209715200, 234881024)    w16    : all weights f16
  //   [234881024, 339738624)    bufS   : f16 102400x512  (h16 / attn / z16)
  //   [339738624, 759169024)    bufL   : f16 102400x2048 (qkv / gelu-out)
  char* ws = (char*)d_ws;
  float*      h    = (float*)ws;
  _Float16*   w16  = (_Float16*)(ws + 209715200ull);
  _Float16*   bufS = (_Float16*)(ws + 234881024ull);
  _Float16*   bufL = (_Float16*)(ws + 339738624ull);

  // weight f16 regions (element offsets within w16)
  const size_t QKV_ALL = 0;            // 4 * 512*1536
  const size_t PRJ_ALL = 3145728;      // 4 * 512*512
  const size_t FF1_ALL = 4194304;      // 4 * 512*2048
  const size_t FF2_ALL = 8388608;      // 4 * 2048*512

  // 1) permute + pos-embed
  k_permute_in<<<(unsigned)((NELEM_X + 255) / 256), 256, 0, stream>>>(x, pos, h);

  // 2) weights -> f16 (counts are multiples of 2048)
  k_cvt_f16<<<3145728 / 2048, 256, 0, stream>>>(qkv_w,  w16 + QKV_ALL, 3145728);
  k_cvt_f16<<<1048576 / 2048, 256, 0, stream>>>(proj_w, w16 + PRJ_ALL, 1048576);
  k_cvt_f16<<<4194304 / 2048, 256, 0, stream>>>(ff_w1,  w16 + FF1_ALL, 4194304);
  k_cvt_f16<<<4194304 / 2048, 256, 0, stream>>>(ff_w2,  w16 + FF2_ALL, 4194304);

  const size_t HN = (size_t)TOKENS * CDIM;  // 52,428,800
  for (int l = 0; l < 4; ++l) {
    const _Float16* w_qkv = w16 + QKV_ALL + (size_t)l * 786432;
    const _Float16* w_prj = w16 + PRJ_ALL + (size_t)l * 262144;
    const _Float16* w_ff1 = w16 + FF1_ALL + (size_t)l * 1048576;
    const _Float16* w_ff2 = w16 + FF2_ALL + (size_t)l * 1048576;

    // h -> f16
    k_cvt_f16<<<(unsigned)(HN / 2048), 256, 0, stream>>>(h, bufS, HN);
    // qkv = h16 @ qkv_w        (102400 x 1536, K=512), f16 out
    k_gemm_wmma<false, false, false, true>
        <<<dim3(800, 24), 256, 0, stream>>>(bufS, w_qkv, nullptr, nullptr, bufL, 512, 1536);
    // windowed attention (double softmax), f16 out
    k_attention<<<NWIN * 8, 256, 0, stream>>>(bufL, bufS);
    // h += attn @ proj_w + proj_b
    k_gemm_wmma<true, false, true, false>
        <<<dim3(800, 8), 256, 0, stream>>>(bufS, w_prj, proj_b + l * 512, h, nullptr, 512, 512);
    // z16 = layernorm(h)
    k_layernorm<<<TOKENS / 8, 256, 0, stream>>>(h, ln_g + l * 512, ln_b + l * 512, bufS);
    // g16 = gelu(z16 @ ff_w1 + b1)   (102400 x 2048, K=512), f16 out
    k_gemm_wmma<true, true, false, true>
        <<<dim3(800, 32), 256, 0, stream>>>(bufS, w_ff1, ff_b1 + l * 2048, nullptr, bufL, 512, 2048);
    // h += g16 @ ff_w2 + b2          (102400 x 512, K=2048)
    k_gemm_wmma<true, false, true, false>
        <<<dim3(800, 8), 256, 0, stream>>>(bufL, w_ff2, ff_b2 + l * 512, h, nullptr, 2048, 512);
  }

  // 3) back to (b, c, n, t)
  k_permute_out<<<(unsigned)((NELEM_X + 255) / 256), 256, 0, stream>>>(h, out);
}
